// DSSMamba2_47296179863956
// MI455X (gfx1250) — matrix operand order repaired and checked by reading, not compiled, not run on hardware
//
#include <hip/hip_runtime.h>
#include <hip/hip_bf16.h>
#include <math.h>
#include <stdint.h>

// ---------------- model constants ----------------
#define LSEQ      1024
#define DMODEL    1024
#define DSTATE    64
#define DINNER    2048
#define NHEADS    32
#define HEADDIM   64
#define CONVDIM   2176      // DINNER + 2*DSTATE
#define DINPROJ   4256      // 2*DINNER + 2*DSTATE + NHEADS
#define EPSV      1e-5f

typedef __bf16 bf16;
typedef bf16  v8bf  __attribute__((ext_vector_type(8)));
typedef bf16  v16bf __attribute__((ext_vector_type(16)));
typedef float v8f   __attribute__((ext_vector_type(8)));

__device__ __forceinline__ float siluf(float x) {
    return x / (1.0f + __expf(-x));
}
__device__ __forceinline__ float softplusf(float x) {
    return (x > 20.0f) ? x : log1pf(__expf(x));
}

// ---------------------------------------------------------------------------
// LDS-staged WMMA GEMM:  C[M,N] = A[M,K] @ W[N,K]^T
// Block tile 128x128, 8 waves (4M x 2N), wave tile 32x64 (2x4 WMMA 16x16).
// Per 32-K chunk both operand tiles are staged in LDS as bf16 (converted once
// on write), double-buffered. Requires M%128==0, K%32==0; N arbitrary mult of 16
// (staged rows clamped, stores guarded).
// ---------------------------------------------------------------------------
#define BM   128
#define BN   128
#define KC   32
#define LDK  40      // padded row stride (bf16 elems) to spread LDS banks

__device__ __forceinline__ void stage16(const float* __restrict__ g, bf16* dst) {
    float4 f0 = *(const float4*)(g + 0);
    float4 f1 = *(const float4*)(g + 4);
    float4 f2 = *(const float4*)(g + 8);
    float4 f3 = *(const float4*)(g + 12);
    v8bf lo, hi;
    lo[0] = (bf16)f0.x; lo[1] = (bf16)f0.y; lo[2] = (bf16)f0.z; lo[3] = (bf16)f0.w;
    lo[4] = (bf16)f1.x; lo[5] = (bf16)f1.y; lo[6] = (bf16)f1.z; lo[7] = (bf16)f1.w;
    hi[0] = (bf16)f2.x; hi[1] = (bf16)f2.y; hi[2] = (bf16)f2.z; hi[3] = (bf16)f2.w;
    hi[4] = (bf16)f3.x; hi[5] = (bf16)f3.y; hi[6] = (bf16)f3.z; hi[7] = (bf16)f3.w;
    *(v8bf*)(dst)     = lo;
    *(v8bf*)(dst + 8) = hi;
}

__global__ __launch_bounds__(256) void wmma_gemm_lds(
    const float* __restrict__ A, const float* __restrict__ W,
    float* __restrict__ C, int M, int N, int K, int flipA)
{
    __shared__ __align__(16) bf16 sA[2][BM * LDK];
    __shared__ __align__(16) bf16 sW[2][BN * LDK];

    const int tid   = threadIdx.x;
    const int lane  = tid & 31;
    const int wave  = tid >> 5;
    const int wm    = wave >> 1;          // 0..3  (M strip of 32)
    const int wn    = wave & 1;           // 0..1  (N strip of 64)
    const int row16 = lane & 15;
    const int half  = lane >> 4;

    const int nbx = (N + BN - 1) / BN;
    const int bx  = blockIdx.x % nbx;
    const int by  = blockIdx.x / nbx;
    const int bm0 = by * BM;
    const int bn0 = bx * BN;

    // staging: thread -> one 16-float segment of one row of each tile
    const int sr = tid >> 1;              // 0..127
    const int sk = (tid & 1) * 16;        // 0 or 16
    int arow = bm0 + sr;
    if (flipA) arow = M - 1 - arow;
    int wrow = bn0 + sr;
    if (wrow > N - 1) wrow = N - 1;       // clamp (garbage cols never stored)
    const float* __restrict__ gA = A + (size_t)arow * K + sk;
    const float* __restrict__ gW = W + (size_t)wrow * K + sk;
    bf16* stA0 = &sA[0][sr * LDK + sk];
    bf16* stA1 = &sA[1][sr * LDK + sk];
    bf16* stW0 = &sW[0][sr * LDK + sk];
    bf16* stW1 = &sW[1][sr * LDK + sk];

    v8f acc[2][4];
    #pragma unroll
    for (int mi = 0; mi < 2; ++mi)
        #pragma unroll
        for (int ni = 0; ni < 4; ++ni) acc[mi][ni] = (v8f){};

    // prime buffer 0
    stage16(gA, stA0);
    stage16(gW, stW0);
    __syncthreads();

    const int nkc = K / KC;
    for (int kc = 0; kc < nkc; ++kc) {
        const int buf = kc & 1;

        // fragments from LDS (all 16B-aligned contiguous chunks)
        v16bf afrag[2], bfrag[4];
        #pragma unroll
        for (int mi = 0; mi < 2; ++mi) {
            const bf16* p = &sA[buf][(wm * 32 + mi * 16 + row16) * LDK];
            v8bf lo = *(const v8bf*)(p + 8 * half);        // k = 8h..8h+7
            v8bf hi = *(const v8bf*)(p + 16 + 8 * half);   // k = 16+8h..+7
            afrag[mi] = __builtin_shufflevector(lo, hi, 0,1,2,3,4,5,6,7,8,9,10,11,12,13,14,15);
        }
        #pragma unroll
        for (int ni = 0; ni < 4; ++ni) {
            const bf16* p = &sW[buf][(wn * 64 + ni * 16 + row16) * LDK];
            v8bf lo = *(const v8bf*)(p + 8 * half);        // k = 8h..8h+7
            v8bf hi = *(const v8bf*)(p + 16 + 8 * half);   // k = 16+8h..+7
            bfrag[ni] = __builtin_shufflevector(lo, hi, 0,1,2,3,4,5,6,7,8,9,10,11,12,13,14,15);
        }

        #pragma unroll
        for (int mi = 0; mi < 2; ++mi)
            #pragma unroll
            for (int ni = 0; ni < 4; ++ni)
                acc[mi][ni] = __builtin_amdgcn_wmma_f32_16x16x32_bf16(
                    false, afrag[mi], false, bfrag[ni], (short)0, acc[mi][ni], false, false);

        // stage next chunk into the other buffer (overlaps with WMMAs above)
        if (kc + 1 < nkc) {
            const int ko = (kc + 1) * KC;
            stage16(gA + ko, (buf ? stA0 : stA1));
            stage16(gW + ko, (buf ? stW0 : stW1));
        }
        __syncthreads();
    }

    // store D: lane%16 = N col, VGPR r -> M = r + 8*half
    #pragma unroll
    for (int mi = 0; mi < 2; ++mi) {
        #pragma unroll
        for (int ni = 0; ni < 4; ++ni) {
            const int col = bn0 + wn * 64 + ni * 16 + row16;
            if (col < N) {
                float* cp = C + (size_t)(bm0 + wm * 32 + mi * 16) * N + col;
                #pragma unroll
                for (int r = 0; r < 8; ++r)
                    cp[(size_t)(r + 8 * half) * N] = acc[mi][ni][r];
            }
        }
    }
}

// ---------------------------------------------------------------------------
// Causal depthwise conv (4 taps) + bias + SiLU over xBC slice of zxbcdt.
// ---------------------------------------------------------------------------
__global__ __launch_bounds__(256) void conv_silu_kernel(
    const float* __restrict__ zx, const float* __restrict__ cw,
    const float* __restrict__ cb, float* __restrict__ out)
{
    int idx = blockIdx.x * 256 + threadIdx.x;
    if (idx >= LSEQ * CONVDIM) return;
    int t = idx / CONVDIM;
    int c = idx - t * CONVDIM;
    float acc = cb[c];
    #pragma unroll
    for (int k = 0; k < 4; ++k) {
        int tt = t - 3 + k;
        if (tt >= 0)
            acc += cw[c * 4 + k] * zx[(size_t)tt * DINPROJ + DINNER + c];
    }
    out[idx] = siluf(acc);
}

// ---------------------------------------------------------------------------
// dt = softplus(raw + dt_bias); dA = exp(dt * (-exp(A_log)))
// ---------------------------------------------------------------------------
__global__ __launch_bounds__(256) void dt_prep_kernel(
    const float* __restrict__ zx, const float* __restrict__ dt_bias,
    const float* __restrict__ A_log, float* __restrict__ dt_out,
    float* __restrict__ dA_out)
{
    int idx = blockIdx.x * 256 + threadIdx.x;
    if (idx >= LSEQ * NHEADS) return;
    int t = idx / NHEADS;
    int h = idx - t * NHEADS;
    float raw = zx[(size_t)t * DINPROJ + (DINPROJ - NHEADS) + h];
    float dt  = softplusf(raw + dt_bias[h]);
    float A   = -__expf(A_log[h]);
    dt_out[idx] = dt;
    dA_out[idx] = __expf(dt * A);
}

// ---------------------------------------------------------------------------
// Sequential SSM scan with async global->LDS double buffering (ASYNCcnt).
// ---------------------------------------------------------------------------
__device__ __forceinline__ void async_lds_load_b32(void* lds_ptr, const float* gptr) {
    unsigned lds_off = (unsigned)(uintptr_t)lds_ptr;   // low 32b of generic = LDS offset
    asm volatile("global_load_async_to_lds_b32 %0, %1, off"
                 :
                 : "v"(lds_off), "v"(gptr)
                 : "memory");
}
__device__ __forceinline__ void wait_async_all() {
    asm volatile("s_wait_asynccnt 0x0" ::: "memory");
}

__global__ __launch_bounds__(64) void scan_kernel(
    const float* __restrict__ conv, const float* __restrict__ dt_arr,
    const float* __restrict__ dA_arr, const float* __restrict__ Dvec,
    float* __restrict__ y)
{
    const int head = blockIdx.x;
    const int p    = threadIdx.x;
    __shared__ float sB[2][DSTATE];
    __shared__ float sC[2][DSTATE];

    float h[DSTATE];
    #pragma unroll
    for (int n = 0; n < DSTATE; ++n) h[n] = 0.0f;
    const float Dh = Dvec[head];

    {   // prime buffer 0 with t=0
        const float* g = conv + DINNER + p;
        async_lds_load_b32(&sB[0][p], g);
        async_lds_load_b32(&sC[0][p], g + DSTATE);
    }

    for (int t = 0; t < LSEQ; ++t) {
        wait_async_all();
        __syncthreads();
        const int cur = t & 1;

        if (t + 1 < LSEQ) {      // uniform branch
            const float* g = conv + (size_t)(t + 1) * CONVDIM + DINNER + p;
            async_lds_load_b32(&sB[cur ^ 1][p], g);
            async_lds_load_b32(&sC[cur ^ 1][p], g + DSTATE);
        }

        const float dt  = dt_arr[t * NHEADS + head];
        const float dA  = dA_arr[t * NHEADS + head];
        const float x   = conv[(size_t)t * CONVDIM + head * HEADDIM + p];
        const float dtx = dt * x;

        float yv = 0.0f;
        #pragma unroll 16
        for (int n = 0; n < DSTATE; ++n) {
            h[n] = h[n] * dA + dtx * sB[cur][n];
            yv  += h[n] * sC[cur][n];
        }
        y[(size_t)t * DINNER + head * HEADDIM + p] = yv + Dh * x;
        __syncthreads();
    }
}

// ---------------------------------------------------------------------------
// yg = rmsnorm(y * silu(z)) * norm_w  (single group of 2048). One block/row.
// ---------------------------------------------------------------------------
__global__ __launch_bounds__(256) void gate_rms_kernel(
    const float* __restrict__ y, const float* __restrict__ zx,
    const float* __restrict__ norm_w, float* __restrict__ yg)
{
    const int t = blockIdx.x;
    __shared__ float red[8];
    float g[8];
    float ss = 0.0f;
    #pragma unroll
    for (int i = 0; i < 8; ++i) {
        int c = threadIdx.x * 8 + i;
        float z  = zx[(size_t)t * DINPROJ + c];
        float gv = y[(size_t)t * DINNER + c] * siluf(z);
        g[i] = gv;
        ss  += gv * gv;
    }
    #pragma unroll
    for (int off = 16; off > 0; off >>= 1)
        ss += __shfl_down(ss, off, 32);
    if ((threadIdx.x & 31) == 0) red[threadIdx.x >> 5] = ss;
    __syncthreads();
    if (threadIdx.x == 0) {
        float s = 0.0f;
        #pragma unroll
        for (int w = 0; w < 8; ++w) s += red[w];
        red[0] = rsqrtf(s / (float)DINNER + EPSV);
    }
    __syncthreads();
    const float r = red[0];
    #pragma unroll
    for (int i = 0; i < 8; ++i) {
        int c = threadIdx.x * 8 + i;
        yg[(size_t)t * DINNER + c] = g[i] * r * norm_w[c];
    }
}

// ---------------------------------------------------------------------------
// tbuf[t][0:1024] = silu(out_f[t]); tbuf[t][1024:2048] = silu(out_b[L-1-t])
// ---------------------------------------------------------------------------
__global__ __launch_bounds__(256) void combine_silu_kernel(
    const float* __restrict__ out_f, const float* __restrict__ out_b,
    float* __restrict__ tbuf)
{
    int idx = blockIdx.x * 256 + threadIdx.x;
    if (idx >= LSEQ * DINNER) return;
    int t = idx / DINNER;
    int c = idx - t * DINNER;
    float v = (c < DMODEL)
                  ? out_f[(size_t)t * DMODEL + c]
                  : out_b[(size_t)(LSEQ - 1 - t) * DMODEL + (c - DMODEL)];
    tbuf[idx] = siluf(v);
}

// ---------------------------------------------------------------------------
extern "C" void kernel_launch(void* const* d_in, const int* in_sizes, int n_in,
                              void* d_out, int out_size, void* d_ws, size_t ws_size,
                              hipStream_t stream)
{
    const float* u        = (const float*)d_in[0];
    const float* W_in_f   = (const float*)d_in[1];
    const float* W_in_b   = (const float*)d_in[2];
    const float* conv_w_f = (const float*)d_in[3];
    const float* conv_b_f = (const float*)d_in[4];
    const float* conv_w_b = (const float*)d_in[5];
    const float* conv_b_b = (const float*)d_in[6];
    const float* dt_b_f   = (const float*)d_in[7];
    const float* dt_b_b   = (const float*)d_in[8];
    const float* A_log_f  = (const float*)d_in[9];
    const float* A_log_b  = (const float*)d_in[10];
    const float* D_f      = (const float*)d_in[11];
    const float* D_b      = (const float*)d_in[12];
    const float* norm_w_f = (const float*)d_in[13];
    const float* norm_w_b = (const float*)d_in[14];
    const float* W_out_f  = (const float*)d_in[15];
    const float* W_out_b  = (const float*)d_in[16];
    const float* W_out    = (const float*)d_in[17];
    float* outp           = (float*)d_out;

    // ---- workspace layout (floats) ----
    float* ws = (float*)d_ws;
    const size_t ZXN = (size_t)LSEQ * DINPROJ;
    const size_t CVN = (size_t)LSEQ * CONVDIM;
    const size_t DTN = (size_t)LSEQ * NHEADS;
    const size_t YN  = (size_t)LSEQ * DINNER;
    float* zx_f   = ws;
    float* zx_b   = zx_f + ZXN;
    float* conv_f = zx_b + ZXN;
    float* conv_b = conv_f + CVN;
    float* dt_f   = conv_b + CVN;
    float* dt_bk  = dt_f + DTN;
    float* dA_f   = dt_bk + DTN;
    float* dA_b   = dA_f + DTN;
    float* y_f    = dA_b + DTN;
    float* y_b    = y_f + YN;
    // after the scan the conv region is dead -> reuse for GEMM outputs + tbuf
    float* out_f  = conv_f;
    float* out_b  = conv_f + (size_t)LSEQ * DMODEL;
    float* tbuf   = conv_f + (size_t)2 * LSEQ * DMODEL;

    // 1) in-projection GEMMs (backward direction reads flipped u)
    {
        int nbx = (DINPROJ + BN - 1) / BN;             // 34
        int blocks = nbx * (LSEQ / BM);                // 272
        wmma_gemm_lds<<<blocks, 256, 0, stream>>>(u, W_in_f, zx_f, LSEQ, DINPROJ, DMODEL, 0);
        wmma_gemm_lds<<<blocks, 256, 0, stream>>>(u, W_in_b, zx_b, LSEQ, DINPROJ, DMODEL, 1);
    }
    // 2) conv + SiLU
    {
        int n = LSEQ * CONVDIM, blocks = (n + 255) / 256;
        conv_silu_kernel<<<blocks, 256, 0, stream>>>(zx_f, conv_w_f, conv_b_f, conv_f);
        conv_silu_kernel<<<blocks, 256, 0, stream>>>(zx_b, conv_w_b, conv_b_b, conv_b);
    }
    // 3) dt / dA
    {
        int n = LSEQ * NHEADS, blocks = (n + 255) / 256;
        dt_prep_kernel<<<blocks, 256, 0, stream>>>(zx_f, dt_b_f, A_log_f, dt_f, dA_f);
        dt_prep_kernel<<<blocks, 256, 0, stream>>>(zx_b, dt_b_b, A_log_b, dt_bk, dA_b);
    }
    // 4) SSM scans (async double-buffered B/C streaming)
    scan_kernel<<<NHEADS, 64, 0, stream>>>(conv_f, dt_f, dA_f, D_f, y_f);
    scan_kernel<<<NHEADS, 64, 0, stream>>>(conv_b, dt_bk, dA_b, D_b, y_b);
    // 5) gate + RMS norm (in place on y)
    gate_rms_kernel<<<LSEQ, 256, 0, stream>>>(y_f, zx_f, norm_w_f, y_f);
    gate_rms_kernel<<<LSEQ, 256, 0, stream>>>(y_b, zx_b, norm_w_b, y_b);
    // 6) per-direction out-projection GEMMs
    {
        int blocks = (DMODEL / BN) * (LSEQ / BM);      // 64
        wmma_gemm_lds<<<blocks, 256, 0, stream>>>(y_f, W_out_f, out_f, LSEQ, DMODEL, DINNER, 0);
        wmma_gemm_lds<<<blocks, 256, 0, stream>>>(y_b, W_out_b, out_b, LSEQ, DMODEL, DINNER, 0);
    }
    // 7) silu(concat(out_f, flip(out_b)))
    {
        int n = LSEQ * DINNER, blocks = (n + 255) / 256;
        combine_silu_kernel<<<blocks, 256, 0, stream>>>(out_f, out_b, tbuf);
    }
    // 8) final projection
    {
        int blocks = (DMODEL / BN) * (LSEQ / BM);      // 64
        wmma_gemm_lds<<<blocks, 256, 0, stream>>>(tbuf, W_out, outp, LSEQ, DMODEL, DINNER, 0);
    }
}
